// PFA_67817533604134
// MI455X (gfx1250) — compile-verified, matrix operand order checked by reading, not run on hardware
//
#include <hip/hip_runtime.h>

#define SS 256
#define BB 16
#define LL 512

typedef __attribute__((ext_vector_type(16))) __bf16 v16bf;
typedef __attribute__((ext_vector_type(8)))  float  v8f;
typedef unsigned short u16;
typedef unsigned int   u32;

__device__ __forceinline__ u16 f32_to_bf16(float f) {
  union { float f; u32 u; } v; v.f = f;
  u32 u = v.u;
  return (u16)((u + 0x7FFFu + ((u >> 16) & 1u)) >> 16);   // round-to-nearest-even
}

// Row scalar c = <z,p*> - 0.5||p*||^2 + 0.5 for sparsemax over a 256-row.
// 256-thread workgroup per row. Hybrid bitonic: intra-wave stages via shuffles,
// only the 6 cross-wave stages (j>=32) touch LDS. sd = 256 floats, wred = 8 floats.
__device__ float sparsemax_row_c(float zval, float* sd, float* wred, int tid) {
  int lane = tid & 31, wave = tid >> 5;

  // ascending bitonic sort of v = -z  (== descending sort of z)
  float v = -zval;
  for (int k = 2; k <= SS; k <<= 1) {
    for (int j = k >> 1; j > 0; j >>= 1) {
      float other;
      if (j >= 32) {
        __syncthreads();
        sd[tid] = v;
        __syncthreads();
        other = sd[tid ^ j];
      } else {
        other = __shfl_xor(v, j, 32);
      }
      bool asc   = ((tid & k) == 0);
      bool lower = ((tid & j) == 0);
      v = (lower == asc) ? fminf(v, other) : fmaxf(v, other);
    }
  }
  float zs = -v;                       // descending sorted value at rank tid

  // inclusive cumsum over ranks: wave shuffle-scan + per-wave offsets
  float sc = zs;
  for (int d = 1; d < 32; d <<= 1) {
    float t = __shfl_up(sc, (unsigned)d, 32);
    if (lane >= d) sc += t;
  }
  __syncthreads();                     // sort reads of sd complete
  if (lane == 31) wred[wave] = sc;
  __syncthreads();
  float base = 0.0f;
#pragma unroll
  for (int w = 0; w < 8; ++w) base += (w < wave) ? wred[w] : 0.0f;
  float zc = sc + base;                // cumsum at rank tid

  // support size via ballot, tau from cumsum at rank kz-1
  bool supp = (1.0f + (float)(tid + 1) * zs) > zc;
  unsigned long long bal = __ballot(supp);
  __syncthreads();                     // wred reads complete
  if (lane == 0) wred[wave] = (float)__popcll(bal);
  sd[tid] = zc;
  __syncthreads();
  int kz = 0;
#pragma unroll
  for (int w = 0; w < 8; ++w) kz += (int)wred[w];
  float tau = (sd[kz - 1] - 1.0f) / (float)kz;

  // c = sum(z*p - 0.5*p^2) + 0.5 : wave reduce + 8 partials
  float p   = fmaxf(zval - tau, 0.0f);
  float acc = zval * p - 0.5f * p * p;
  for (int d = 16; d > 0; d >>= 1) acc += __shfl_xor(acc, d, 32);
  __syncthreads();                     // wred reads complete
  if (lane == 0) wred[wave] = acc;
  __syncthreads();
  float c = 0.5f;
#pragma unroll
  for (int w = 0; w < 8; ++w) c += wred[w];
  return c;
}

// --- Phase 1 kernels -------------------------------------------------------

// T rows: c -> cT[q*256+x]; fused  A[x][q][next] = bf16(exp(-T[q][x][next]))
__global__ void __launch_bounds__(256) kTab(const float* __restrict__ T,
                                            float* __restrict__ cT,
                                            u16* __restrict__ Abf) {
  __shared__ float sd[SS];
  __shared__ float wred[8];
  int r = blockIdx.x, tid = threadIdx.x;
  int q = r >> 8, x = r & 255;
  float z = T[(size_t)r * SS + tid];
  float c = sparsemax_row_c(z, sd, wred, tid);
  Abf[(size_t)x * (SS * SS) + q * SS + tid] = f32_to_bf16(__expf(-z));
  if (tid == 0) cT[q * SS + x] = c;
}

__global__ void __launch_bounds__(256) kErow(const float* __restrict__ E,
                                             float* __restrict__ cE) {
  __shared__ float sd[SS];
  __shared__ float wred[8];
  int q = blockIdx.x, tid = threadIdx.x;
  float z = E[q * SS + tid];
  float c = sparsemax_row_c(z, sd, wred, tid);
  if (tid == 0) cE[q] = c;
}

__global__ void __launch_bounds__(256) kAlOm(const float* __restrict__ alpha,
                                             const float* __restrict__ omega,
                                             float* __restrict__ LAl,
                                             float* __restrict__ LOm) {
  __shared__ float sd[SS];
  __shared__ float wred[8];
  int tid = threadIdx.x;
  const float* src = blockIdx.x ? omega : alpha;
  float z = src[tid];
  float c = sparsemax_row_c(z, sd, wred, tid);
  (blockIdx.x ? LOm : LAl)[tid] = c - z;
}

// bias[x*256+q] = cE[q] - E[q,x] + cT[q,x]
__global__ void __launch_bounds__(256) kBias(const float* __restrict__ E,
                                             const float* __restrict__ cT,
                                             const float* __restrict__ cE,
                                             float* __restrict__ bias) {
  int x = blockIdx.x, q = threadIdx.x;
  bias[x * SS + q] = cE[q] - E[q * SS + x] + cT[q * SS + x];
}

// --- Phase 2: persistent backward scan, one workgroup per batch element ----
// chart update: c[q] <- bias_x[q] + cm + log( (A_x @ exp(c-cm))[q] )  via WMMA bf16
// 3 barriers per step; 32-lane reductions via shuffles (wave32).

__global__ void __launch_bounds__(256) kScan(const int* __restrict__ xs,
                                             const u16* __restrict__ Abf,
                                             const float* __restrict__ bias,
                                             const float* __restrict__ LAl,
                                             const float* __restrict__ LOm,
                                             float* __restrict__ out) {
  __shared__ float c[SS];
  __shared__ float red[8];
  __shared__ __align__(16) u16 s16[SS];

  int b = blockIdx.x, tid = threadIdx.x;
  int lane = tid & 31, wave = tid >> 5;
  int half = lane >> 4;                 // 0: K 0-7/16-23 ; 1: K 8-15/24-31
  int kb   = half * 8;

  c[tid] = LOm[tid];
  __syncthreads();

  for (int t = LL - 1; t >= 0; --t) {
    int x = xs[b * LL + t];
    if (x != 0) {
      const u16* Ax = Abf + (size_t)x * (SS * SS);
      // fully cover next step's 128KB matrix: 4 prefetches/thread @128B stride
      if (t > 0) {
        int xn = xs[b * LL + t - 1];
        if (xn != 0) {
          const char* pn = (const char*)(Abf + (size_t)xn * (SS * SS)) + tid * 512;
          __builtin_prefetch(pn,       0, 1);
          __builtin_prefetch(pn + 128, 0, 1);
          __builtin_prefetch(pn + 256, 0, 1);
          __builtin_prefetch(pn + 384, 0, 1);
        }
      }
      // cm = max_q c[q] : wave shuffle max + 8 partials (1 barrier)
      float cv = c[tid];
      float m = cv;
      for (int d = 16; d > 0; d >>= 1) m = fmaxf(m, __shfl_xor(m, d, 32));
      if (lane == 0) red[wave] = m;
      __syncthreads();
      float cm = red[0];
#pragma unroll
      for (int w = 1; w < 8; ++w) cm = fmaxf(cm, red[w]);

      // s = exp(c - cm) in bf16 (WMMA B operand, replicated across N)
      s16[tid] = f32_to_bf16(__expf(cv - cm));
      __syncthreads();

      v8f acc0 = {}, acc1 = {};
      int q0 = wave * 32 + (lane & 15);     // A-row (M) for tile 0
      int q1 = q0 + 16;                     // tile 1
      const u16* ap0 = Ax + q0 * SS;
      const u16* ap1 = Ax + q1 * SS;
#pragma unroll
      for (int kt = 0; kt < 8; ++kt) {
        int ks = kt * 32 + kb;
        union { v16bf v; uint4 u[2]; } bf, a0, a1;
        bf.u[0] = *(const uint4*)(&s16[ks]);
        bf.u[1] = *(const uint4*)(&s16[ks + 16]);
        a0.u[0] = *(const uint4*)(ap0 + ks);
        a0.u[1] = *(const uint4*)(ap0 + ks + 16);
        a1.u[0] = *(const uint4*)(ap1 + ks);
        a1.u[1] = *(const uint4*)(ap1 + ks + 16);
        acc0 = __builtin_amdgcn_wmma_f32_16x16x32_bf16(false, a0.v, false, bf.v,
                                                       (short)0, acc0, false, false);
        acc1 = __builtin_amdgcn_wmma_f32_16x16x32_bf16(false, a1.v, false, bf.v,
                                                       (short)0, acc1, false, false);
      }
      // D layout: lane group holds rows (half*8 + r); all 16 columns identical.
      if ((lane & 15) == 0) {
        const float* bx = bias + x * SS;
#pragma unroll
        for (int r = 0; r < 8; ++r) {
          int qa = wave * 32 + half * 8 + r;
          int qb = qa + 16;
          c[qa] = bx[qa] + cm + __logf(acc0[r]);
          c[qb] = bx[qb] + cm + __logf(acc1[r]);
        }
      }
      __syncthreads();
    }
  }

  // per_seq[b] = logsumexp_q(LAl[q] + c[q]);  sum over b via atomic
  float v = LAl[tid] + c[tid];
  float m = v;
  for (int d = 16; d > 0; d >>= 1) m = fmaxf(m, __shfl_xor(m, d, 32));
  if (lane == 0) red[wave] = m;
  __syncthreads();
  float gm = red[0];
#pragma unroll
  for (int w = 1; w < 8; ++w) gm = fmaxf(gm, red[w]);
  float e = __expf(v - gm);
  for (int d = 16; d > 0; d >>= 1) e += __shfl_xor(e, d, 32);
  __syncthreads();                     // red reads complete
  if (lane == 0) red[wave] = e;
  __syncthreads();
  if (tid == 0) {
    float s = red[0];
#pragma unroll
    for (int w = 1; w < 8; ++w) s += red[w];
    atomicAdd(out, gm + __logf(s));
  }
}

// --- Host launcher ---------------------------------------------------------

extern "C" void kernel_launch(void* const* d_in, const int* in_sizes, int n_in,
                              void* d_out, int out_size, void* d_ws, size_t ws_size,
                              hipStream_t stream) {
  const int*   xs    = (const int*)d_in[0];
  const float* alpha = (const float*)d_in[1];
  const float* omega = (const float*)d_in[2];
  const float* E     = (const float*)d_in[3];
  const float* T     = (const float*)d_in[4];
  float*       out   = (float*)d_out;

  // workspace carve-up (~33 MB)
  char* p = (char*)d_ws;
  u16*   Abf  = (u16*)p;            p += (size_t)SS * SS * SS * sizeof(u16);   // 32 MB
  float* cT   = (float*)p;          p += (size_t)SS * SS * sizeof(float);
  float* bias = (float*)p;          p += (size_t)SS * SS * sizeof(float);
  float* cE   = (float*)p;          p += SS * sizeof(float);
  float* LAl  = (float*)p;          p += SS * sizeof(float);
  float* LOm  = (float*)p;          p += SS * sizeof(float);

  hipMemsetAsync(out, 0, out_size * sizeof(float), stream);

  kAlOm<<<2, SS, 0, stream>>>(alpha, omega, LAl, LOm);
  kErow<<<SS, SS, 0, stream>>>(E, cE);
  kTab <<<SS * SS, SS, 0, stream>>>(T, cT, Abf);
  kBias<<<SS, SS, 0, stream>>>(E, cT, cE, bias);
  kScan<<<BB, SS, 0, stream>>>(xs, Abf, bias, LAl, LOm, out);
}
